// _CLAM_Base_29609504539084
// MI455X (gfx1250) — matrix operand order repaired and checked by reading, not compiled, not run on hardware
//
#include <hip/hip_runtime.h>
#include <hip/hip_bf16.h>
#include <math.h>

typedef __bf16 bf16;
typedef __attribute__((ext_vector_type(16))) __bf16 v16bf;
typedef __attribute__((ext_vector_type(8)))  __bf16 bf16x8;
typedef __attribute__((ext_vector_type(8)))  float  v8f;

#define N_ROWS 65536
#define L_DIM  1024
#define D1     512
#define D2     256
#define KS     8

#define G1_BM 64
#define G1_LDS_STRIDE 1032   /* bf16 elems/row: 1024 + 8 pad -> 2064B, 16B aligned, 4-bank row skew */
#define G2_BM 32
#define G2_STRIDE 520        /* 512 + 8 pad -> 1040B */
#define P_BLOCKS 256
#define P_ROWS (N_ROWS / P_BLOCKS)

static __device__ __forceinline__ uint32_t pack_bf16(float a, float b) {
  union { bf16 h[2]; uint32_t u; } p;
  p.h[0] = (bf16)a; p.h[1] = (bf16)b;
  return p.u;
}

// ---------------------------------------------------------------- weight cvt
__global__ void k_f32_to_bf16(const float* __restrict__ in, bf16* __restrict__ outp, int n) {
  int i = blockIdx.x * blockDim.x + threadIdx.x;
  if (i < n) outp[i] = (bf16)in[i];
}

// ---------------------------------------------------------------- GEMM1: X = relu(h @ W1^T + b1)  (bf16 out)
__global__ __launch_bounds__(256) void k_gemm1(const float* __restrict__ h,
                                               const bf16*  __restrict__ W1b,
                                               const float* __restrict__ b1,
                                               bf16* __restrict__ Xb) {
  __shared__ bf16 sA[G1_BM * G1_LDS_STRIDE];
  const int t = threadIdx.x;
  const int rowbase = blockIdx.x * G1_BM;

  // stage 64 rows of h -> bf16 LDS (coalesced float4 reads; VALU path needed for f32->bf16 cvt)
  for (int i = 0; i < (G1_BM * L_DIM) / (256 * 4); ++i) {
    int flat = i * (256 * 4) + t * 4;
    int r = flat >> 10;
    int c = flat & (L_DIM - 1);
    const float4 v = *(const float4*)(h + (size_t)(rowbase + r) * L_DIM + c);
    uint32_t* dst = (uint32_t*)(sA + r * G1_LDS_STRIDE + c);
    dst[0] = pack_bf16(v.x, v.y);
    dst[1] = pack_bf16(v.z, v.w);
  }
  __syncthreads();

  const int wave = t >> 5;
  const int lane = t & 31;
  const int half = lane >> 4;
  const int n    = lane & 15;
  const int wcol = wave * 64;   // each wave owns a disjoint 64-col strip

  const v8f vzero = {0.f,0.f,0.f,0.f,0.f,0.f,0.f,0.f};
  v8f acc[4][4];
  #pragma unroll
  for (int bd = 0; bd < 4; ++bd)
    #pragma unroll
    for (int ct = 0; ct < 4; ++ct) acc[bd][ct] = vzero;

  union Frag { v16bf v; bf16x8 h8[2]; };

  for (int k0 = 0; k0 < L_DIM; k0 += 32) {
    Frag a[4];
    #pragma unroll
    for (int bd = 0; bd < 4; ++bd) {
      const bf16* ap = sA + (bd * 16 + n) * G1_LDS_STRIDE + k0 + half * 16;
      a[bd].h8[0] = *(const bf16x8*)ap;          // K = k0 + half*16 .. +7
      a[bd].h8[1] = *(const bf16x8*)(ap + 8);    // K = k0+half*16+8 .. +15
    }
    #pragma unroll
    for (int ct = 0; ct < 4; ++ct) {
      Frag b;
      const bf16* bp = W1b + (size_t)(wcol + ct * 16 + n) * L_DIM + k0 + half * 16;
      b.h8[0] = *(const bf16x8*)bp;              // K = k0+half*16 .. +7
      b.h8[1] = *(const bf16x8*)(bp + 8);        // K = k0+half*16+8 .. +15
      #pragma unroll
      for (int bd = 0; bd < 4; ++bd)
        acc[bd][ct] = __builtin_amdgcn_wmma_f32_16x16x32_bf16(
            false, a[bd].v, false, b.v, (short)0, acc[bd][ct], false, false);
    }
  }

  // epilogue: bias + relu, store bf16 X
  #pragma unroll
  for (int ct = 0; ct < 4; ++ct) {
    const int col = wcol + ct * 16 + n;
    const float bias = b1[col];
    #pragma unroll
    for (int bd = 0; bd < 4; ++bd) {
      #pragma unroll
      for (int r = 0; r < 8; ++r) {
        int row = rowbase + bd * 16 + half * 8 + r;
        float v = fmaxf(acc[bd][ct][r] + bias, 0.f);
        Xb[(size_t)row * D1 + col] = (bf16)v;
      }
    }
  }
}

// ---------------------------------------------------------------- GEMM2: raw attention scores
__global__ __launch_bounds__(256) void k_attn(const bf16* __restrict__ Xb,
                                              const bf16* __restrict__ Wab,
                                              const bf16* __restrict__ Wbb,
                                              const float* __restrict__ ba,
                                              const float* __restrict__ bb,
                                              const float* __restrict__ Wc,
                                              const float* __restrict__ bc,
                                              float* __restrict__ scores) {
  __shared__ bf16  sX[G2_BM * G2_STRIDE];
  __shared__ float sred[8][16];
  const int t = threadIdx.x;
  const int rowbase = blockIdx.x * G2_BM;

  // stage 32 bf16 rows of X via the CDNA5 async copy engine (memory -> LDS, no VGPR round-trip).
  // 32 rows * 1024B = 2048 b128 chunks; 256 threads -> 8 chunks each.
  for (int i = 0; i < 8; ++i) {
    int fu = i * 256 + t;
    int r  = fu >> 6;               // 64 chunks per row
    int cc = fu & 63;
    const bf16* gp = Xb + (size_t)(rowbase + r) * D1 + cc * 8;
    uint32_t ldsoff = (uint32_t)(uintptr_t)(sX + r * G2_STRIDE + cc * 8);
    asm volatile("global_load_async_to_lds_b128 %0, %1, off"
                 :: "v"(ldsoff), "v"(gp) : "memory");
  }
  asm volatile("s_wait_asynccnt 0x0" ::: "memory");
  __syncthreads();

  const int wave = t >> 5, lane = t & 31, half = lane >> 4, n = lane & 15;
  const int bd    = wave >> 2;          // row band 0..1
  const int strip = (wave & 3) * 64;    // col strip

  const v8f vzero = {0.f,0.f,0.f,0.f,0.f,0.f,0.f,0.f};
  v8f accA[4], accG[4];
  #pragma unroll
  for (int ct = 0; ct < 4; ++ct) { accA[ct] = vzero; accG[ct] = vzero; }

  union Frag { v16bf v; bf16x8 h8[2]; };

  for (int k0 = 0; k0 < D1; k0 += 32) {
    Frag af;
    const bf16* ap = sX + (bd * 16 + n) * G2_STRIDE + k0 + half * 16;
    af.h8[0] = *(const bf16x8*)ap;
    af.h8[1] = *(const bf16x8*)(ap + 8);
    #pragma unroll
    for (int ct = 0; ct < 4; ++ct) {
      const size_t boff = (size_t)(strip + ct * 16 + n) * D1 + k0 + half * 16;
      Frag bA, bB;
      bA.h8[0] = *(const bf16x8*)(Wab + boff);
      bA.h8[1] = *(const bf16x8*)(Wab + boff + 8);
      accA[ct] = __builtin_amdgcn_wmma_f32_16x16x32_bf16(
          false, af.v, false, bA.v, (short)0, accA[ct], false, false);
      bB.h8[0] = *(const bf16x8*)(Wbb + boff);
      bB.h8[1] = *(const bf16x8*)(Wbb + boff + 8);
      accG[ct] = __builtin_amdgcn_wmma_f32_16x16x32_bf16(
          false, af.v, false, bB.v, (short)0, accG[ct], false, false);
    }
  }

  // fused gated-attention epilogue: tanh(a)*sigmoid(g)*Wc, reduce over D2
  float part[8] = {0.f,0.f,0.f,0.f,0.f,0.f,0.f,0.f};
  #pragma unroll
  for (int ct = 0; ct < 4; ++ct) {
    const int col = strip + ct * 16 + n;
    const float bav = ba[col], bbv = bb[col], wcv = Wc[col];
    #pragma unroll
    for (int r = 0; r < 8; ++r) {
      float av = tanhf(accA[ct][r] + bav);
      float gv = 1.f / (1.f + expf(-(accG[ct][r] + bbv)));
      part[r] += av * gv * wcv;
    }
  }
  #pragma unroll
  for (int r = 0; r < 8; ++r) {
    float v = part[r];
    for (int m = 1; m < 16; m <<= 1) v += __shfl_xor(v, m, 16);
    part[r] = v;
  }
  if (n == 0) {
    #pragma unroll
    for (int r = 0; r < 8; ++r) sred[wave][half * 8 + r] = part[r];
  }
  __syncthreads();
  if (t < G2_BM) {
    int row = t;
    int w0  = (row >> 4) * 4;
    int ir  = row & 15;
    float s = sred[w0][ir] + sred[w0 + 1][ir] + sred[w0 + 2][ir] + sred[w0 + 3][ir];
    scores[rowbase + row] = s + bc[0];
  }
}

// ---------------------------------------------------------------- softmax stats + top/bottom-8 (single block, deterministic)
__global__ __launch_bounds__(1024) void k_softmax_topk(const float* __restrict__ scores,
                                                       float* __restrict__ statsF) {
  __shared__ float sv[1024];
  __shared__ int   si[1024];
  __shared__ int   pickedT[KS];
  __shared__ int   pickedB[KS];
  __shared__ float s_smax;
  const int t = threadIdx.x;
  int* statsI = (int*)statsF;

  // max
  float mx = -INFINITY;
  for (int i = t; i < N_ROWS; i += 1024) mx = fmaxf(mx, scores[i]);
  sv[t] = mx; __syncthreads();
  for (int s = 512; s > 0; s >>= 1) { if (t < s) sv[t] = fmaxf(sv[t], sv[t + s]); __syncthreads(); }
  if (t == 0) s_smax = sv[0];
  __syncthreads();
  const float smax = s_smax;
  __syncthreads();

  // sum exp
  float sum = 0.f;
  for (int i = t; i < N_ROWS; i += 1024) sum += expf(scores[i] - smax);
  sv[t] = sum; __syncthreads();
  for (int s = 512; s > 0; s >>= 1) { if (t < s) sv[t] += sv[t + s]; __syncthreads(); }
  if (t == 0) { statsF[0] = smax; statsF[1] = sv[0]; }
  __syncthreads();

  // top-8 (argmax, tie -> lower index)
  for (int iter = 0; iter < KS; ++iter) {
    float bv = -INFINITY; int bix = 0x7fffffff;
    for (int i = t; i < N_ROWS; i += 1024) {
      bool skip = false;
      for (int j = 0; j < iter; ++j) if (pickedT[j] == i) skip = true;
      if (skip) continue;
      float v = scores[i];
      if (v > bv || (v == bv && i < bix)) { bv = v; bix = i; }
    }
    __syncthreads();
    sv[t] = bv; si[t] = bix; __syncthreads();
    for (int s = 512; s > 0; s >>= 1) {
      if (t < s) {
        if (sv[t + s] > sv[t] || (sv[t + s] == sv[t] && si[t + s] < si[t])) {
          sv[t] = sv[t + s]; si[t] = si[t + s];
        }
      }
      __syncthreads();
    }
    if (t == 0) { pickedT[iter] = si[0]; statsI[2 + iter] = si[0]; }
    __syncthreads();
  }

  // bottom-8 (argmin, tie -> lower index)
  for (int iter = 0; iter < KS; ++iter) {
    float bv = INFINITY; int bix = 0x7fffffff;
    for (int i = t; i < N_ROWS; i += 1024) {
      bool skip = false;
      for (int j = 0; j < iter; ++j) if (pickedB[j] == i) skip = true;
      if (skip) continue;
      float v = scores[i];
      if (v < bv || (v == bv && i < bix)) { bv = v; bix = i; }
    }
    __syncthreads();
    sv[t] = bv; si[t] = bix; __syncthreads();
    for (int s = 512; s > 0; s >>= 1) {
      if (t < s) {
        if (sv[t + s] < sv[t] || (sv[t + s] == sv[t] && si[t + s] < si[t])) {
          sv[t] = sv[t + s]; si[t] = si[t + s];
        }
      }
      __syncthreads();
    }
    if (t == 0) { pickedB[iter] = si[0]; statsI[10 + iter] = si[0]; }
    __syncthreads();
  }
}

// ---------------------------------------------------------------- pooling partials: sum_n exp(s-smax)*x  (deterministic)
__global__ __launch_bounds__(256) void k_pool_partial(const bf16* __restrict__ Xb,
                                                      const float* __restrict__ scores,
                                                      const float* __restrict__ statsF,
                                                      float* __restrict__ partials) {
  const int t = threadIdx.x, b = blockIdx.x;
  const float smax = statsF[0];
  float a0 = 0.f, a1 = 0.f;
  for (int r = 0; r < P_ROWS; ++r) {
    int row = b * P_ROWS + r;
    float w = expf(scores[row] - smax);
    union { uint32_t u; bf16 h[2]; } q;
    q.u = ((const uint32_t*)(Xb + (size_t)row * D1))[t];
    a0 += w * (float)q.h[0];
    a1 += w * (float)q.h[1];
  }
  partials[(size_t)b * D1 + 2 * t]     = a0;
  partials[(size_t)b * D1 + 2 * t + 1] = a1;
}

__global__ __launch_bounds__(256) void k_pool_reduce(const float* __restrict__ partials,
                                                     const float* __restrict__ statsF,
                                                     float* __restrict__ M) {
  const int t = threadIdx.x;
  const float invZ = 1.f / statsF[1];
  float a0 = 0.f, a1 = 0.f;
  for (int b = 0; b < P_BLOCKS; ++b) {
    a0 += partials[(size_t)b * D1 + 2 * t];
    a1 += partials[(size_t)b * D1 + 2 * t + 1];
  }
  M[2 * t]     = a0 * invZ;
  M[2 * t + 1] = a1 * invZ;
}

// ---------------------------------------------------------------- instance CE + bag logits
__global__ __launch_bounds__(256) void k_final(const bf16* __restrict__ Xb,
                                               const float* __restrict__ statsF,
                                               const float* __restrict__ Wi,
                                               const float* __restrict__ bi,
                                               const float* __restrict__ Wcls,
                                               const float* __restrict__ bcls,
                                               const float* __restrict__ M,
                                               const int* __restrict__ label,
                                               float* __restrict__ out) {
  __shared__ float lg[16][2][2];
  __shared__ float lgt[2];
  const int t = threadIdx.x;
  const int* statsI = (const int*)statsF;

  if (t < 64) {
    int row = t >> 2, i = (t >> 1) & 1, j = t & 1;
    int id = (row < KS) ? statsI[2 + row] : statsI[10 + (row - KS)];
    const bf16*  xr = Xb + (size_t)id * D1;
    const float* wv = Wi + ((size_t)i * 2 + j) * D1;
    float s = bi[i * 2 + j];
    for (int d = 0; d < D1; ++d) s += (float)xr[d] * wv[d];
    lg[row][i][j] = s;
  } else if (t < 66) {
    int c = t - 64;
    float s = bcls[c];
    for (int d = 0; d < D1; ++d) s += M[d] * Wcls[(size_t)c * D1 + d];
    lgt[c] = s;
  }
  __syncthreads();

  if (t == 0) {
    int lbl = label[0];
    float loss = 0.f;
    for (int i = 0; i < 2; ++i) {
      float ce = 0.f;
      for (int row = 0; row < 16; ++row) {
        int tgt = (row < KS) ? 1 : 0;
        float l0 = lg[row][i][0], l1 = lg[row][i][1];
        float mm  = fmaxf(l0, l1);
        float lse = mm + logf(expf(l0 - mm) + expf(l1 - mm));
        ce += -((tgt ? l1 : l0) - lse);
      }
      ce *= (1.f / 16.f);
      loss += ((i == lbl) ? 1.f : 0.f) * ce;
    }
    out[0] = lgt[0];
    out[1] = lgt[1];
    out[2] = loss;
  }
}

// ---------------------------------------------------------------- launch
extern "C" void kernel_launch(void* const* d_in, const int* in_sizes, int n_in,
                              void* d_out, int out_size, void* d_ws, size_t ws_size,
                              hipStream_t stream) {
  (void)in_sizes; (void)n_in; (void)out_size; (void)ws_size;
  const float* h    = (const float*)d_in[0];
  const float* W1   = (const float*)d_in[1];
  const float* b1   = (const float*)d_in[2];
  const float* Wa   = (const float*)d_in[3];
  const float* ba   = (const float*)d_in[4];
  const float* Wb   = (const float*)d_in[5];
  const float* bb   = (const float*)d_in[6];
  const float* Wc   = (const float*)d_in[7];
  const float* bc   = (const float*)d_in[8];
  const float* Wi   = (const float*)d_in[9];
  const float* bi   = (const float*)d_in[10];
  const float* Wcls = (const float*)d_in[11];
  const float* bcls = (const float*)d_in[12];
  const int*   label= (const int*)d_in[13];
  float* out = (float*)d_out;

  char* ws = (char*)d_ws;
  size_t off = 0;
  bf16* W1b = (bf16*)(ws + off); off += (size_t)D1 * L_DIM * 2;
  bf16* Wab = (bf16*)(ws + off); off += (size_t)D2 * D1 * 2;
  bf16* Wbb = (bf16*)(ws + off); off += (size_t)D2 * D1 * 2;
  bf16* Xb  = (bf16*)(ws + off); off += (size_t)N_ROWS * D1 * 2;
  float* scores   = (float*)(ws + off); off += (size_t)N_ROWS * 4;
  float* statsF   = (float*)(ws + off); off += 256;
  float* partials = (float*)(ws + off); off += (size_t)P_BLOCKS * D1 * 4;
  float* M        = (float*)(ws + off); off += (size_t)D1 * 4;

  k_f32_to_bf16<<<(D1 * L_DIM + 255) / 256, 256, 0, stream>>>(W1, W1b, D1 * L_DIM);
  k_f32_to_bf16<<<(D2 * D1 + 255) / 256, 256, 0, stream>>>(Wa, Wab, D2 * D1);
  k_f32_to_bf16<<<(D2 * D1 + 255) / 256, 256, 0, stream>>>(Wb, Wbb, D2 * D1);
  k_gemm1<<<N_ROWS / G1_BM, 256, 0, stream>>>(h, W1b, b1, Xb);
  k_attn<<<N_ROWS / G2_BM, 256, 0, stream>>>(Xb, Wab, Wbb, ba, bb, Wc, bc, scores);
  k_softmax_topk<<<1, 1024, 0, stream>>>(scores, statsF);
  k_pool_partial<<<P_BLOCKS, 256, 0, stream>>>(Xb, scores, statsF, partials);
  k_pool_reduce<<<1, 256, 0, stream>>>(partials, statsF, M);
  k_final<<<1, 256, 0, stream>>>(Xb, statsF, Wi, bi, Wcls, bcls, M, label, out);
}
